// OCAP_60327110640023
// MI455X (gfx1250) — matrix operand order repaired and checked by reading, not compiled, not run on hardware
//
#include <hip/hip_runtime.h>
#include <hip/hip_bf16.h>

typedef __bf16 bf16_t;
typedef __attribute__((ext_vector_type(16))) __bf16 v16bf;
typedef __attribute__((ext_vector_type(8)))  float        v8f;
typedef __attribute__((ext_vector_type(8)))  unsigned int v8u;

constexpr int Bb = 32, Nn = 128, SEQ = 336, PRED = 96, Tt = 432;
constexpr int DIN = 16, DEMB = 32, HID = 64, Gg = 8, KK = 25, ENDC = 34;
constexpr int SEQP = 352;                       // SEQ padded to multiple of 32 (WMMA K)
constexpr long O_ATTF  = (long)Bb * Nn * SEQ;   // out_former size
constexpr long O_AFTER = O_ATTF + (long)Bb * SEQ;
constexpr long O_ATTA  = O_AFTER + (long)Bb * Nn * PRED;

// ---------------------------------------------------------------- WMMA helpers
__device__ inline v8f wmma_bf16(v16bf a, v16bf b, v8f c) {
  return __builtin_amdgcn_wmma_f32_16x16x32_bf16(false, a, false, b, (short)0, c,
                                                 false, false);
}

// A tile 16x32 bf16 from row-major [M][K] memory (contiguous k order, matching load_tileB):
// lanes 0-15: row=lane, K 0..15 in v0..7
// lanes 16-31: row=lane-16, K 16..31 in v0..7
__device__ inline v16bf load_tileA(const bf16_t* base, int lda) {
  int lane = threadIdx.x & 31;
  int row  = lane & 15;
  int kh   = (lane >> 4) << 3;  // dword offset 0 or 8
  const unsigned int* p =
      reinterpret_cast<const unsigned int*>(base + (size_t)row * lda);
  v8u u;
#pragma unroll
  for (int j = 0; j < 4; ++j) u[j] = p[kh + j];
#pragma unroll
  for (int j = 0; j < 4; ++j) u[4 + j] = p[kh + 4 + j];
  return __builtin_bit_cast(v16bf, u);
}

// B tile 32x16 bf16 from memory holding B^T row-major as [N][K]:
// lanes 0-15: col=lane, K=0..15 in v0..7; lanes 16-31: col=lane-16, K=16..31
__device__ inline v16bf load_tileB(const bf16_t* base, int ldb) {
  int lane = threadIdx.x & 31;
  int col  = lane & 15;
  int kh   = (lane >> 4) << 3;  // dword offset 0 or 8
  const unsigned int* p =
      reinterpret_cast<const unsigned int*>(base + (size_t)col * ldb);
  v8u u;
#pragma unroll
  for (int j = 0; j < 8; ++j) u[j] = p[kh + j];
  return __builtin_bit_cast(v16bf, u);
}

// dy B tile: [16 timesteps][K=16 f32] -> bf16, K zero-padded to 32 (hi lanes = 0)
__device__ inline v16bf load_dyB(const float* base) {
  int lane = threadIdx.x & 31;
  int col  = lane & 15;
  bool hi  = lane >= 16;
  const float* p = base + col * DIN;
  v16bf v;
#pragma unroll
  for (int j = 0; j < 16; ++j) v[j] = hi ? (bf16_t)0.f : (bf16_t)p[j];
  return v;
}

// ---------------------------------------------------------------- prep weights
__global__ void k_prep(const float* w_seas, const float* w_trend, const float* w_res,
                       const float* w1_end, const float* dy_w,
                       bf16_t* wseas_bf, bf16_t* wtrend_bf, bf16_t* wres_bf,
                       bf16_t* w1_bf, bf16_t* dyw_bf) {
  const int M2 = Gg * PRED * SEQP;   // 270336 per seq-weight
  const int M5 = Gg * HID * 64;      // 32768
  const int M6 = DEMB * 32;          // 1024
  const int total = 3 * M2 + M5 + M6;
  for (int i = blockIdx.x * blockDim.x + threadIdx.x; i < total;
       i += gridDim.x * blockDim.x) {
    int idx = i;
    if (idx < 3 * M2) {
      int w = idx / M2, r = idx % M2;
      int s = r % SEQP, gp = r / SEQP;
      const float* src = (w == 0) ? w_seas : (w == 1) ? w_trend : w_res;
      bf16_t* dst      = (w == 0) ? wseas_bf : (w == 1) ? wtrend_bf : wres_bf;
      dst[r] = (s < SEQ) ? (bf16_t)src[gp * SEQ + s] : (bf16_t)0.f;
      continue;
    }
    idx -= 3 * M2;
    if (idx < M5) {
      int k = idx % 64, gh = idx / 64;
      w1_bf[idx] = (k < ENDC) ? (bf16_t)w1_end[gh * ENDC + k] : (bf16_t)0.f;
      continue;
    }
    idx -= M5;
    int k = idx % 32, e = idx / 32;
    dyw_bf[idx] = (k < DIN) ? (bf16_t)dy_w[e * DIN + k] : (bf16_t)0.f;
  }
}

__global__ void k_count(const int* gi, float* cnt) {
  __shared__ int c[Gg];
  if (threadIdx.x < Gg) c[threadIdx.x] = 0;
  __syncthreads();
  if (threadIdx.x < Nn) atomicAdd(&c[gi[threadIdx.x]], 1);
  __syncthreads();
  if (threadIdx.x < Gg) cnt[threadIdx.x] = (float)c[threadIdx.x];
}

// -------------------------------------------------- series decomposition (K=25)
__global__ void k_decomp(const float* x, bf16_t* seas_bf, bf16_t* trend_bf) {
  int b = blockIdx.x / Nn, n = blockIdx.x % Nn;
  __shared__ float xs[SEQ];
  int tid = threadIdx.x;
  if (tid < SEQ) xs[tid] = x[(size_t)(b * Nn + n) * SEQ + tid];
  __syncthreads();
  if (tid < SEQP) {
    float tr = 0.f, se = 0.f;
    if (tid < SEQ) {
      float s = 0.f;
#pragma unroll
      for (int j = 0; j < KK; ++j) {
        int u = tid + j - (KK - 1) / 2;
        u = u < 0 ? 0 : (u > SEQ - 1 ? SEQ - 1 : u);
        s += xs[u];
      }
      tr = s * (1.f / KK);
      se = xs[tid] - tr;
    }
    seas_bf[(size_t)(b * Nn + n) * SEQP + tid]  = (bf16_t)se;
    trend_bf[(size_t)(b * Nn + n) * SEQP + tid] = (bf16_t)tr;
  }
}

// ------------------------------- deterministic per-(b,t) stats over the N nodes
__global__ void k_colstats(const float* x, const bf16_t* trend_bf, const int* gi,
                           const float* cnt, float* ginf, float* meanH) {
  int b = blockIdx.x / SEQ, t = blockIdx.x % SEQ;
  int n = threadIdx.x;  // 128
  __shared__ float sx[Nn], st[Nn];
  __shared__ int sg[Nn];
  float xv = x[(size_t)(b * Nn + n) * SEQ + t];
  float tv = (float)trend_bf[(size_t)(b * Nn + n) * SEQP + t];
  sx[n] = xv; st[n] = tv; sg[n] = gi[n];
  __syncthreads();
  if (n < Gg) {                       // per-group mean of x (serial = deterministic)
    float s = 0.f;
    for (int i = 0; i < Nn; ++i)
      if (sg[i] == n) s += sx[i];
    ginf[((size_t)n * Bb + b) * SEQ + t] = s / cnt[n];
  }
  __syncthreads();
  for (int s2 = 64; s2 > 0; s2 >>= 1) {
    if (n < s2) { sx[n] += sx[n + s2]; st[n] += st[n + s2]; }
    __syncthreads();
  }
  if (n == 0) {
    meanH[(b * Tt + t) * ENDC + 0] = (sx[0] - st[0]) * (1.f / Nn);  // seasonal mean
    meanH[(b * Tt + t) * ENDC + 1] = st[0] * (1.f / Nn);            // trend mean
  }
}

__global__ void k_ginbf(const float* ginf, bf16_t* gin_bf) {
  int i = blockIdx.x * blockDim.x + threadIdx.x;
  if (i >= Gg * Bb * SEQP) return;
  int s = i % SEQP, gb = i / SEQP;
  gin_bf[i] = (s < SEQ) ? (bf16_t)ginf[gb * SEQ + s] : (bf16_t)0.f;
}

// -------------------------------------- g_out = w_res x g_in + b_res (WMMA bf16)
__global__ void k_gout(const bf16_t* wres_bf, const bf16_t* gin_bf,
                       const float* b_res, float* gout) {
  int id = blockIdx.x;
  int g = id / 12, r = id % 12, mt = r >> 1, nt = r & 1;
  const bf16_t* A  = wres_bf + (size_t)(g * PRED + mt * 16) * SEQP;
  const bf16_t* Bm = gin_bf + (size_t)(g * Bb + nt * 16) * SEQP;
  v8f acc = {};
  for (int k0 = 0; k0 < SEQP; k0 += 32)
    acc = wmma_bf16(load_tileA(A + k0, SEQP), load_tileB(Bm + k0, SEQP), acc);
  int lane = threadIdx.x & 31;
  int col = lane & 15, hoff = (lane >> 4) * 8;
  int bb = nt * 16 + col;
#pragma unroll
  for (int j = 0; j < 8; ++j) {
    int p = mt * 16 + j + hoff;
    gout[(g * Bb + bb) * PRED + p] = acc[j] + b_res[g * PRED + p];
  }
}

// -------------------- per-node seas/trend heads: [96x352] x [352x32] (WMMA bf16)
__global__ void k_heads(const bf16_t* wseas_bf, const bf16_t* wtrend_bf,
                        const bf16_t* seas_bf, const bf16_t* trend_bf,
                        const float* b_seas, const float* b_trend, const int* gi,
                        float* seasout, float* trendout) {
  int id = blockIdx.x;
  int which = id & 1; id >>= 1;
  int nt = id & 1;    id >>= 1;
  int mt = id % 6, n = id / 6;
  int g = gi[n];
  const bf16_t* A  = (which ? wtrend_bf : wseas_bf) + (size_t)(g * PRED + mt * 16) * SEQP;
  const bf16_t* Bm = (which ? trend_bf : seas_bf) + ((size_t)(nt * 16) * Nn + n) * SEQP;
  const float* bias = which ? b_trend : b_seas;
  float* outp       = which ? trendout : seasout;
  v8f acc = {};
  for (int k0 = 0; k0 < SEQP; k0 += 32)
    acc = wmma_bf16(load_tileA(A + k0, SEQP), load_tileB(Bm + k0, Nn * SEQP), acc);
  int lane = threadIdx.x & 31;
  int col = lane & 15, hoff = (lane >> 4) * 8;
  int bb = nt * 16 + col;
#pragma unroll
  for (int j = 0; j < 8; ++j) {
    int p = mt * 16 + j + hoff;
    outp[((size_t)bb * Nn + n) * PRED + p] = acc[j] + bias[g * PRED + p];
  }
}

__global__ void k_meanaft(const float* seasout, const float* trendout, float* meanH) {
  int b = blockIdx.x / PRED, p = blockIdx.x % PRED;
  int n = threadIdx.x;
  __shared__ float s0[Nn], s1[Nn];
  s0[n] = seasout[((size_t)b * Nn + n) * PRED + p];
  s1[n] = trendout[((size_t)b * Nn + n) * PRED + p];
  __syncthreads();
  for (int s = 64; s > 0; s >>= 1) {
    if (n < s) { s0[n] += s0[n + s]; s1[n] += s1[n + s]; }
    __syncthreads();
  }
  if (n == 0) {
    meanH[(b * Tt + SEQ + p) * ENDC + 0] = s0[0] * (1.f / Nn);
    meanH[(b * Tt + SEQ + p) * ENDC + 1] = s1[0] * (1.f / Nn);
  }
}

// ----------------------------- mean over nodes of relu(dy @ dyW^T + b) (GRU in)
__global__ void k_meanemb(const float* dy, const float* dy_w, const float* dy_b,
                          float* meanH) {
  int b = blockIdx.x / Tt, t = blockIdx.x % Tt;
  int e = threadIdx.x & 31, ns = threadIdx.x >> 5;
  float w[DIN];
#pragma unroll
  for (int j = 0; j < DIN; ++j) w[j] = dy_w[e * DIN + j];
  float bias = dy_b[e];
  float acc = 0.f;
  for (int n = ns; n < Nn; n += 8) {
    const float* d = dy + ((size_t)(b * Nn + n) * Tt + t) * DIN;
    float s = bias;
#pragma unroll
    for (int j = 0; j < DIN; ++j) s += d[j] * w[j];
    acc += s > 0.f ? s : 0.f;
  }
  __shared__ float red[256];
  red[threadIdx.x] = acc;
  __syncthreads();
  if (ns < 4) red[threadIdx.x] += red[threadIdx.x + 128];
  __syncthreads();
  if (ns < 2) red[threadIdx.x] += red[threadIdx.x + 64];
  __syncthreads();
  if (ns == 0)
    meanH[(b * Tt + t) * ENDC + 2 + e] = (red[e] + red[e + 32]) * (1.f / Nn);
}

// ---------- fused dy-embedding + end MLP + residual, one wave per (b,n,16 steps)
__global__ void k_endmlp(const float* dy, const bf16_t* dyw_bf, const float* dy_b,
                         const bf16_t* seas_bf, const bf16_t* trend_bf,
                         const float* seasout, const float* trendout,
                         const bf16_t* w1_bf, const float* b1_end,
                         const float* w2_end, const float* b2_end,
                         const float* ginf, const float* gout, const int* gi,
                         float* out) {
  int id = blockIdx.x;
  int tt = id % 27; id /= 27;
  int n = id % Nn, b = id / Nn;
  int g = gi[n];
  int t0 = tt * 16;
  int lane = threadIdx.x & 31;
  __shared__ __align__(16) bf16_t HT[16 * 64];  // H^T tile: [16 t][64 padded channels]
  {
    unsigned int* hz = (unsigned int*)HT;
#pragma unroll
    for (int i = 0; i < 16; ++i) hz[lane + i * 32] = 0u;
  }
  __syncthreads();
  // dy embedding: 2 WMMAs (DEMB=32 rows, K=16 padded to 32)
  const float* dyp = dy + ((size_t)(b * Nn + n) * Tt + t0) * DIN;
  v16bf bt = load_dyB(dyp);
  v8f zc = {};
  v8f e0 = wmma_bf16(load_tileA(dyw_bf, 32), bt, zc);
  v8f e1 = wmma_bf16(load_tileA(dyw_bf + 16 * 32, 32), bt, zc);
  int col = lane & 15, hoff = (lane >> 4) * 8;
#pragma unroll
  for (int j = 0; j < 8; ++j) {
    int ea = j + hoff;
    float v = e0[j] + dy_b[ea];       v = v > 0.f ? v : 0.f;
    HT[col * 64 + 2 + ea] = (bf16_t)v;
    int eb = 16 + ea;
    float u = e1[j] + dy_b[eb];       u = u > 0.f ? u : 0.f;
    HT[col * 64 + 2 + eb] = (bf16_t)u;
  }
  if (lane < 16) {
    int t = t0 + lane;
    bf16_t c0, c1;
    if (t < SEQ) {
      c0 = seas_bf[(size_t)(b * Nn + n) * SEQP + t];
      c1 = trend_bf[(size_t)(b * Nn + n) * SEQP + t];
    } else {
      int p = t - SEQ;
      c0 = (bf16_t)seasout[((size_t)b * Nn + n) * PRED + p];
      c1 = (bf16_t)trendout[((size_t)b * Nn + n) * PRED + p];
    }
    HT[lane * 64 + 0] = c0;
    HT[lane * 64 + 1] = c1;
  }
  __syncthreads();
  // Z = W1_g (64 x 64padded) x H : 4 M-tiles x 2 K-steps = 8 WMMAs
  v8f z[4];
#pragma unroll
  for (int mt = 0; mt < 4; ++mt) {
    v8f a = {};
#pragma unroll
    for (int ks = 0; ks < 2; ++ks)
      a = wmma_bf16(load_tileA(w1_bf + (size_t)(g * HID + mt * 16) * 64 + ks * 32, 64),
                    load_tileB(HT + ks * 32, 64), a);
    z[mt] = a;
  }
  // out[t] = w2 . relu(z + b1) + b2 + residual
  float part = 0.f;
#pragma unroll
  for (int mt = 0; mt < 4; ++mt) {
#pragma unroll
    for (int j = 0; j < 8; ++j) {
      int h = mt * 16 + j + hoff;
      float v = z[mt][j] + b1_end[g * HID + h];
      v = v > 0.f ? v : 0.f;
      part += w2_end[g * HID + h] * v;
    }
  }
  part += __shfl_xor(part, 16);
  if (lane < 16) {
    int t = t0 + lane;
    float o = part + b2_end[g];
    if (t < SEQ) {
      o += ginf[(g * Bb + b) * SEQ + t];
      out[((size_t)(b * Nn + n)) * SEQ + t] = o;          // out_former
    } else {
      int p = t - SEQ;
      o += gout[(g * Bb + b) * PRED + p];
      out[O_AFTER + ((size_t)(b * Nn + n)) * PRED + p] = o;  // out_after
    }
  }
}

// ---------------------------------------------------- GRU + attention projection
__global__ void k_gru(const float* meanH, const float* w_ih, const float* w_hh,
                      const float* b_ih, const float* b_hh,
                      const float* w_ap, const float* b_ap, float* out) {
  int b = blockIdx.x;
  int h = threadIdx.x;  // 64 threads = HID
  __shared__ float s_whh[3 * HID * HID];  // 48 KB, LDS-resident recurrent weights
  __shared__ float hs[HID], xs[ENDC], red[HID];
  for (int i = h; i < 3 * HID * HID; i += HID) s_whh[i] = w_hh[i];
  float wr[ENDC], wz[ENDC], wn[ENDC];
#pragma unroll
  for (int j = 0; j < ENDC; ++j) {
    wr[j] = w_ih[h * ENDC + j];
    wz[j] = w_ih[(h + HID) * ENDC + j];
    wn[j] = w_ih[(h + 2 * HID) * ENDC + j];
  }
  float bir = b_ih[h], biz = b_ih[h + HID], bin = b_ih[h + 2 * HID];
  float bhr = b_hh[h], bhz = b_hh[h + HID], bhn = b_hh[h + 2 * HID];
  float wap = w_ap[h], bap = b_ap[0];
  for (int seg = 0; seg < 2; ++seg) {
    int ts = seg ? SEQ : 0, te = seg ? Tt : SEQ;
    float hv = 0.f;
    __syncthreads();
    hs[h] = 0.f;
    __syncthreads();
    for (int t = ts; t < te; ++t) {
      if (h < ENDC) xs[h] = meanH[(b * Tt + t) * ENDC + h];
      __syncthreads();
      float ar = bir, az = biz, an = bin;
#pragma unroll
      for (int j = 0; j < ENDC; ++j) {
        float xv = xs[j];
        ar += wr[j] * xv; az += wz[j] * xv; an += wn[j] * xv;
      }
      float hr = bhr, hz = bhz, hn = bhn;
      for (int j = 0; j < HID; ++j) {
        float hj = hs[j];
        hr += s_whh[h * HID + j] * hj;
        hz += s_whh[(h + HID) * HID + j] * hj;
        hn += s_whh[(h + 2 * HID) * HID + j] * hj;
      }
      float r  = 1.f / (1.f + __expf(-(ar + hr)));
      float zz = 1.f / (1.f + __expf(-(az + hz)));
      float ng = tanhf(an + r * hn);
      hv = (1.f - zz) * ng + zz * hv;
      __syncthreads();
      hs[h] = hv;
      red[h] = hv * wap;
      __syncthreads();
      for (int s = 32; s > 0; s >>= 1) {
        if (h < s) red[h] += red[h + s];
        __syncthreads();
      }
      if (h == 0) {
        float a = 1.f / (1.f + __expf(-(red[0] + bap)));
        if (seg == 0) out[O_ATTF + b * SEQ + t] = a;
        else          out[O_ATTA + b * PRED + (t - SEQ)] = a;
      }
      __syncthreads();
    }
  }
}

// --------------------------------------------------------------------- launcher
extern "C" void kernel_launch(void* const* d_in, const int* in_sizes, int n_in,
                              void* d_out, int out_size, void* d_ws, size_t ws_size,
                              hipStream_t stream) {
  (void)in_sizes; (void)n_in; (void)out_size; (void)ws_size;
  const float* x       = (const float*)d_in[0];
  const float* dy      = (const float*)d_in[1];
  const int*   gi      = (const int*)d_in[2];
  const float* dy_w    = (const float*)d_in[3];
  const float* dy_b    = (const float*)d_in[4];
  const float* w_seas  = (const float*)d_in[5];
  const float* b_seas  = (const float*)d_in[6];
  const float* w_trend = (const float*)d_in[7];
  const float* b_trend = (const float*)d_in[8];
  const float* w_res   = (const float*)d_in[9];
  const float* b_res   = (const float*)d_in[10];
  const float* w1_end  = (const float*)d_in[11];
  const float* b1_end  = (const float*)d_in[12];
  const float* w2_end  = (const float*)d_in[13];
  const float* b2_end  = (const float*)d_in[14];
  const float* w_ih    = (const float*)d_in[15];
  const float* w_hh    = (const float*)d_in[16];
  const float* b_ih    = (const float*)d_in[17];
  const float* b_hh    = (const float*)d_in[18];
  const float* w_ap    = (const float*)d_in[19];
  const float* b_ap    = (const float*)d_in[20];
  float* out = (float*)d_out;

  char* ws = (char*)d_ws;
  size_t off = 0;
  auto bump = [&](size_t bytes) -> void* {
    void* p = ws + off;
    off += (bytes + 255) & ~(size_t)255;
    return p;
  };
  bf16_t* wseas_bf  = (bf16_t*)bump((size_t)Gg * PRED * SEQP * 2);
  bf16_t* wtrend_bf = (bf16_t*)bump((size_t)Gg * PRED * SEQP * 2);
  bf16_t* wres_bf   = (bf16_t*)bump((size_t)Gg * PRED * SEQP * 2);
  bf16_t* w1_bf     = (bf16_t*)bump((size_t)Gg * HID * 64 * 2);
  bf16_t* dyw_bf    = (bf16_t*)bump((size_t)DEMB * 32 * 2);
  bf16_t* seas_bf   = (bf16_t*)bump((size_t)Bb * Nn * SEQP * 2);
  bf16_t* trend_bf  = (bf16_t*)bump((size_t)Bb * Nn * SEQP * 2);
  bf16_t* gin_bf    = (bf16_t*)bump((size_t)Gg * Bb * SEQP * 2);
  float*  ginf      = (float*)bump((size_t)Gg * Bb * SEQ * 4);
  float*  gout      = (float*)bump((size_t)Gg * Bb * PRED * 4);
  float*  seasout   = (float*)bump((size_t)Bb * Nn * PRED * 4);
  float*  trendout  = (float*)bump((size_t)Bb * Nn * PRED * 4);
  float*  meanH     = (float*)bump((size_t)Bb * Tt * ENDC * 4);
  float*  cnt       = (float*)bump((size_t)Gg * 4);

  const int prepTotal = 3 * Gg * PRED * SEQP + Gg * HID * 64 + DEMB * 32;
  k_prep<<<(prepTotal + 255) / 256, 256, 0, stream>>>(
      w_seas, w_trend, w_res, w1_end, dy_w,
      wseas_bf, wtrend_bf, wres_bf, w1_bf, dyw_bf);
  k_count<<<1, 128, 0, stream>>>(gi, cnt);
  k_decomp<<<Bb * Nn, 384, 0, stream>>>(x, seas_bf, trend_bf);
  k_colstats<<<Bb * SEQ, 128, 0, stream>>>(x, trend_bf, gi, cnt, ginf, meanH);
  k_ginbf<<<(Gg * Bb * SEQP + 255) / 256, 256, 0, stream>>>(ginf, gin_bf);
  k_gout<<<Gg * 12, 32, 0, stream>>>(wres_bf, gin_bf, b_res, gout);
  k_heads<<<Nn * 24, 32, 0, stream>>>(wseas_bf, wtrend_bf, seas_bf, trend_bf,
                                      b_seas, b_trend, gi, seasout, trendout);
  k_meanaft<<<Bb * PRED, 128, 0, stream>>>(seasout, trendout, meanH);
  k_meanemb<<<Bb * Tt, 256, 0, stream>>>(dy, dy_w, dy_b, meanH);
  k_endmlp<<<Bb * Nn * 27, 32, 0, stream>>>(
      dy, dyw_bf, dy_b, seas_bf, trend_bf, seasout, trendout,
      w1_bf, b1_end, w2_end, b2_end, ginf, gout, gi, out);
  k_gru<<<Bb, 64, 0, stream>>>(meanH, w_ih, w_hh, b_ih, b_hh, w_ap, b_ap, out);
}
